// EdgeConv_43619687858604
// MI455X (gfx1250) — matrix operand order repaired and checked by reading, not compiled, not run on hardware
//
#include <hip/hip_runtime.h>
#include <hip/hip_bf16.h>

typedef __attribute__((ext_vector_type(16))) _Float16 v16h;
typedef __attribute__((ext_vector_type(8)))  _Float16 v8h;
typedef __attribute__((ext_vector_type(8)))  float    v8f;
typedef __attribute__((ext_vector_type(2)))  float    v2f;

#define BB   8
#define NN   4096
#define DD   64
#define KNB  10
#define ROWS (BB * NN)          // 32768
#define KCH  128                // KNN key chunk (async-staged in LDS)
#define DLD  (KCH + 4)          // padded dist ld (floats)
#define AQLD 68                 // padded query-tile ld (floats)
#define KLD  68                 // padded key-buffer ld (floats)
#define SEG  8                  // selection segments per row

// ---------------- WMMA wrappers ----------------
__device__ __forceinline__ v8f wmma_f16(v16h a, v16h b, v8f c) {
  return __builtin_amdgcn_wmma_f32_16x16x32_f16(false, a, false, b, (short)0, c, false, false);
}
__device__ __forceinline__ v8f wmma_f32k4(v2f a, v2f b, v8f c) {
  return __builtin_amdgcn_wmma_f32_16x16x4_f32(false, a, false, b, (short)0, c, false, false);
}

// ---------------- CDNA5 async global->LDS copy (ASYNCcnt-tracked) ----------------
__device__ __forceinline__ void async_ld_b128(unsigned lds_byte_addr, unsigned long long gaddr) {
  asm volatile("global_load_async_to_lds_b128 %0, %1, off"
               :: "v"(lds_byte_addr), "v"(gaddr) : "memory");
}
__device__ __forceinline__ void wait_async0() {
  asm volatile("s_wait_asynccnt 0x0" ::: "memory");
}
__device__ __forceinline__ void wait_ds0() {
  asm volatile("s_wait_dscnt 0x0" ::: "memory");
}

// A-operand (f16 16x32): lane holds row, halves at [k0+8h, +8) and [k0+16+8h, +8)
__device__ __forceinline__ v16h loadA16(const _Float16* p) {
  v8h lo = *(const v8h*)(p);
  v8h hi = *(const v8h*)(p + 16);
  v16h a;
#pragma unroll
  for (int i = 0; i < 8; ++i) { a[i] = lo[i]; a[i + 8] = hi[i]; }
  return a;
}
// B-operand (f16 32x16): lane holds col, halves at [k0+8h, +8) and [k0+16+8h, +8)
__device__ __forceinline__ v16h loadB16(const _Float16* p) {
  v8h lo = *(const v8h*)(p);
  v8h hi = *(const v8h*)(p + 16);
  v16h a;
#pragma unroll
  for (int i = 0; i < 8; ++i) { a[i] = lo[i]; a[i + 8] = hi[i]; }
  return a;
}

// ---------------- 1) weight prep: f16 transposed [out][in] ----------------
__global__ void k_prep(const float* __restrict__ W1, const float* __restrict__ W2,
                       const float* __restrict__ W3, const float* __restrict__ Wg,
                       _Float16* __restrict__ t1, _Float16* __restrict__ t2,
                       _Float16* __restrict__ t3, _Float16* __restrict__ tg) {
  int i = blockIdx.x * blockDim.x + threadIdx.x;
  if (i < 16384) {                 // W1: 128x128
    int in = i >> 7, out = i & 127;
    t1[out * 128 + in] = (_Float16)W1[in * 128 + out];
  } else if (i < 24576) {          // W2: 128x64
    int j = i - 16384; int in = j >> 6, out = j & 63;
    t2[out * 128 + in] = (_Float16)W2[in * 64 + out];
  } else if (i < 28672) {          // W3: 64x64
    int j = i - 24576; int in = j >> 6, out = j & 63;
    t3[out * 64 + in] = (_Float16)W3[in * 64 + out];
  } else if (i < 36864) {          // Wg: 128x64
    int j = i - 28672; int in = j >> 6, out = j & 63;
    tg[out * 128 + in] = (_Float16)Wg[in * 64 + out];
  }
}

// ---------------- 2) zero global-max buffer ----------------
__global__ void k_zero(unsigned int* __restrict__ g) {
  int i = blockIdx.x * blockDim.x + threadIdx.x;
  if (i < BB * DD) g[i] = 0u;
}

// ---------------- 3) BN partial sums (deterministic) ----------------
__global__ void k_bnstats(const float* __restrict__ x, float* __restrict__ partial) {
  __shared__ float sS[256], sQ[256];
  int t = threadIdx.x;
  int c = t & 63, rg = t >> 6;
  float s = 0.f, q = 0.f;
  int rowbase = blockIdx.x * 128;
  for (int rr = rg; rr < 128; rr += 4) {
    float v = x[(size_t)(rowbase + rr) * DD + c];
    s += v; q += v * v;
  }
  sS[t] = s; sQ[t] = q;
  __syncthreads();
  if (t < 64) {
    float ts = 0.f, tq = 0.f;
#pragma unroll
    for (int gI = 0; gI < 4; ++gI) { ts += sS[gI * 64 + t]; tq += sQ[gI * 64 + t]; }
    partial[blockIdx.x * 128 + t]      = ts;
    partial[blockIdx.x * 128 + 64 + t] = tq;
  }
}

// ---------------- 4) reduce partials ----------------
__global__ void k_bnreduce(const float* __restrict__ partial, float* __restrict__ stats) {
  int c = threadIdx.x;  // 0..127
  float s = 0.f;
  for (int b = 0; b < 256; ++b) s += partial[b * 128 + c];
  stats[c] = s;
}

// ---------------- 5) normalize + row squared norms ----------------
__global__ void k_normalize(const float* __restrict__ x, const float* __restrict__ gamma,
                            const float* __restrict__ beta, const float* __restrict__ stats,
                            float* __restrict__ xn, float* __restrict__ sq) {
  __shared__ float scale[64], shift[64];
  int t = threadIdx.x;
  if (t < 64) {
    float mu  = stats[t] * (1.f / ROWS);
    float var = stats[64 + t] * (1.f / ROWS) - mu * mu;
    float rs  = rsqrtf(var + 1e-5f);
    float sc  = gamma[t] * rs;
    scale[t] = sc;
    shift[t] = beta[t] - mu * sc;
  }
  __syncthreads();
  int row = blockIdx.x * blockDim.x + t;
  const float* xr = x + (size_t)row * DD;
  float* xo = xn + (size_t)row * DD;
  float acc = 0.f;
#pragma unroll 8
  for (int c = 0; c < DD; ++c) {
    float v = xr[c] * scale[c] + shift[c];
    xo[c] = v;
    acc += v * v;
  }
  sq[row] = acc;
}

// ---------------- 6) KNN: async-staged keys + fp32 WMMA gram + top-10 ----------------
__global__ void __launch_bounds__(256) k_knn(const float* __restrict__ xn,
                                             const float* __restrict__ sq,
                                             int* __restrict__ idxout) {
  extern __shared__ char smem[];
  float* Aq     = (float*)smem;                        // 16*AQLD
  float* sqq    = Aq + 16 * AQLD;                      // 16
  float* dist   = sqq + 16;                            // 16*DLD
  float* candV  = dist + 16 * DLD;                     // 16*SEG*KNB
  int*   candI  = (int*)(candV + 16 * SEG * KNB);      // 16*SEG*KNB
  float* runV   = (float*)(candI + 16 * SEG * KNB);    // 160
  int*   runI   = (int*)(runV + 16 * KNB);             // 160
  float* keybuf = (float*)(runI + 16 * KNB);           // KCH*KLD (async-staged)

  int t = threadIdx.x;
  int lane = t & 31, wave = t >> 5;
  int kh = lane >> 4, l15 = lane & 15;
  int b  = blockIdx.x >> 8;
  int qbase = (blockIdx.x & 255) * 16;
  const float* xb  = xn + (size_t)b * NN * DD;
  const float* sqb = sq + (size_t)b * NN;

  for (int i = t; i < 16 * DD; i += 256) {
    int r = i >> 6, c = i & 63;
    Aq[r * AQLD + c] = xb[(size_t)(qbase + r) * DD + c];
  }
  if (t < 16) sqq[t] = sqb[qbase + t];
  if (t < 16 * KNB) { runV[t] = 3.0e38f; runI[t] = 0; }

  // each wave owns keybuf rows [wave*16, wave*16+16): prefetch chunk 0
  float* kdst = keybuf + (wave * 16) * KLD;
  {
    const float* src = xb;  // chunk 0, keys wave*16..+16
#pragma unroll
    for (int j = 0; j < 8; ++j) {
      int cid = lane + 32 * j;                 // 256 x 16B chunks for 16 rows
      int r = cid >> 4, o = (cid & 15) * 4;    // o in floats
      async_ld_b128((unsigned)(size_t)(kdst + r * KLD + o),
                    (unsigned long long)(size_t)(src + (size_t)(wave * 16 + r) * DD + o));
    }
  }
  __syncthreads();

  for (int ch = 0; ch < NN / KCH; ++ch) {
    int kbase = ch * KCH;
    // ---- gram for this wave's 16-key tile (keys kbase + wave*16 ..) ----
    wait_async0();
    {
      const float* qrow = Aq + l15 * AQLD;
      const float* krow = keybuf + (wave * 16 + l15) * KLD;
      v8f a0 = {}, a1 = {};       // two independent K-chains for ILP
#pragma unroll
      for (int ks = 0; ks < 32; ks += 4) {
        v2f av, bv;
        av[0] = qrow[ks + 2 * kh]; av[1] = qrow[ks + 2 * kh + 1];
        bv[0] = krow[ks + 2 * kh]; bv[1] = krow[ks + 2 * kh + 1];
        a0 = wmma_f32k4(av, bv, a0);
      }
#pragma unroll
      for (int ks = 32; ks < 64; ks += 4) {
        v2f av, bv;
        av[0] = qrow[ks + 2 * kh]; av[1] = qrow[ks + 2 * kh + 1];
        bv[0] = krow[ks + 2 * kh]; bv[1] = krow[ks + 2 * kh + 1];
        a1 = wmma_f32k4(av, bv, a1);
      }
      float sk = sqb[kbase + wave * 16 + l15];
      int cloc = wave * 16 + l15;
#pragma unroll
      for (int v = 0; v < 8; ++v) {
        int qr = v + 8 * kh;
        dist[qr * DLD + cloc] = sqq[qr] + sk - 2.f * (a0[v] + a1[v]);
      }
    }
    // ---- prefetch next chunk into this wave's region (overlaps selection) ----
    wait_ds0();   // our ds reads of keybuf are done before async engine overwrites
    if (ch + 1 < NN / KCH) {
      const float* src = xb + (size_t)(kbase + KCH) * DD;
#pragma unroll
      for (int j = 0; j < 8; ++j) {
        int cid = lane + 32 * j;
        int r = cid >> 4, o = (cid & 15) * 4;
        async_ld_b128((unsigned)(size_t)(kdst + r * KLD + o),
                      (unsigned long long)(size_t)(src + (size_t)(wave * 16 + r) * DD + o));
      }
    }
    __syncthreads();
    // ---- per-thread local top-10 over 16 columns (threads 0..127) ----
    if (t < 16 * SEG) {
      int r = t >> 3, s = t & (SEG - 1);
      float bvv[KNB]; int bii[KNB];
#pragma unroll
      for (int i = 0; i < KNB; ++i) { bvv[i] = 3.0e38f; bii[i] = 0; }
      const float* dr = dist + r * DLD;
      for (int c = s * (KCH / SEG); c < (s + 1) * (KCH / SEG); ++c) {
        float d = dr[c];
        if (d < bvv[KNB - 1]) {
          int gi = kbase + c;
          int j = KNB - 1;
          while (j > 0 && bvv[j - 1] > d) { bvv[j] = bvv[j - 1]; bii[j] = bii[j - 1]; --j; }
          bvv[j] = d; bii[j] = gi;
        }
      }
#pragma unroll
      for (int i = 0; i < KNB; ++i) {
        candV[(r * SEG + s) * KNB + i] = bvv[i];
        candI[(r * SEG + s) * KNB + i] = bii[i];
      }
    }
    __syncthreads();
    if (t < 16) {   // merge SEG sorted candidate lists into running top-10
      int r = t;
      float rv[KNB]; int ri[KNB];
#pragma unroll
      for (int i = 0; i < KNB; ++i) { rv[i] = runV[r * KNB + i]; ri[i] = runI[r * KNB + i]; }
      for (int s = 0; s < SEG; ++s) {
        for (int jj = 0; jj < KNB; ++jj) {
          float d = candV[(r * SEG + s) * KNB + jj];
          if (d >= rv[KNB - 1]) break;   // candidate list sorted ascending
          int gi = candI[(r * SEG + s) * KNB + jj];
          int j = KNB - 1;
          while (j > 0 && rv[j - 1] > d) { rv[j] = rv[j - 1]; ri[j] = ri[j - 1]; --j; }
          rv[j] = d; ri[j] = gi;
        }
      }
#pragma unroll
      for (int i = 0; i < KNB; ++i) { runV[r * KNB + i] = rv[i]; runI[r * KNB + i] = ri[i]; }
    }
    __syncthreads();
  }
  if (t < 16 * KNB) {
    int r = t / KNB, j = t % KNB;
    idxout[((size_t)b * NN + qbase + r) * KNB + j] = runI[r * KNB + j];
  }
}

// ---------------- 7) edge build + 3 WMMA GEMMs + neighbor max ----------------
#define ELD 136   // padded ld for 128-wide f16 LDS tiles
#define HLD 72    // padded ld for 64-wide f16 LDS tiles
__global__ void __launch_bounds__(256) k_mlp(const float* __restrict__ xn, const int* __restrict__ idx,
                     const _Float16* __restrict__ wt1, const _Float16* __restrict__ wt2,
                     const _Float16* __restrict__ wt3,
                     const float* __restrict__ b1, const float* __restrict__ b2,
                     const float* __restrict__ b3, float* __restrict__ vert) {
  extern __shared__ char smem[];
  _Float16* edge = (_Float16*)smem;           // 80 x ELD (reused as h3 f32 later)
  _Float16* h1   = edge + 80 * ELD;           // 80 x ELD
  _Float16* h2   = h1 + 80 * ELD;             // 80 x HLD
  int*      idxL = (int*)(h2 + 80 * HLD);     // 80
  float*    h3   = (float*)smem;              // 80 x 64 f32, aliases edge

  int t = threadIdx.x;
  int lane = t & 31, wave = t >> 5;
  int kh = lane >> 4, l15 = lane & 15;
  int pbase = blockIdx.x * 8;                 // global point row
  int b = pbase >> 12;

  if (t < 80) idxL[t] = b * NN + idx[(size_t)pbase * KNB + t];
  __syncthreads();

  for (int e = t; e < 80 * 128; e += 256) {
    int row = e >> 7, col = e & 127;
    int p = pbase + row / KNB;
    float v;
    if (col < 64) v = xn[(size_t)p * DD + col];
    else {
      int nb = idxL[row];
      v = xn[(size_t)p * DD + (col - 64)] - xn[(size_t)nb * DD + (col - 64)];
    }
    edge[row * ELD + col] = (_Float16)v;
  }
  __syncthreads();

  // GEMM1: [80x128]@[128x128] -> h1, 5 mtiles x 8 ntiles
  for (int tile = wave; tile < 40; tile += 8) {
    int mt = tile >> 3, nt = tile & 7;
    int ocol = nt * 16 + l15;
    v8f acc = {};
#pragma unroll
    for (int kc = 0; kc < 128; kc += 32) {
      v16h a  = loadA16(edge + (mt * 16 + l15) * ELD + kc + 8 * kh);
      v16h bm = loadB16(wt1 + ocol * 128 + kc + 8 * kh);
      acc = wmma_f16(a, bm, acc);
    }
    float bias = b1[ocol];
#pragma unroll
    for (int v = 0; v < 8; ++v)
      h1[(mt * 16 + v + 8 * kh) * ELD + ocol] = (_Float16)fmaxf(acc[v] + bias, 0.f);
  }
  __syncthreads();

  // GEMM2: [80x128]@[128x64] -> h2, 5 x 4 tiles
  for (int tile = wave; tile < 20; tile += 8) {
    int mt = tile >> 2, nt = tile & 3;
    int ocol = nt * 16 + l15;
    v8f acc = {};
#pragma unroll
    for (int kc = 0; kc < 128; kc += 32) {
      v16h a  = loadA16(h1 + (mt * 16 + l15) * ELD + kc + 8 * kh);
      v16h bm = loadB16(wt2 + ocol * 128 + kc + 8 * kh);
      acc = wmma_f16(a, bm, acc);
    }
    float bias = b2[ocol];
#pragma unroll
    for (int v = 0; v < 8; ++v)
      h2[(mt * 16 + v + 8 * kh) * HLD + ocol] = (_Float16)fmaxf(acc[v] + bias, 0.f);
  }
  __syncthreads();

  // GEMM3: [80x64]@[64x64] -> h3 (f32, aliases edge region)
  for (int tile = wave; tile < 20; tile += 8) {
    int mt = tile >> 2, nt = tile & 3;
    int ocol = nt * 16 + l15;
    v8f acc = {};
#pragma unroll
    for (int kc = 0; kc < 64; kc += 32) {
      v16h a  = loadA16(h2 + (mt * 16 + l15) * HLD + kc + 8 * kh);
      v16h bm = loadB16(wt3 + ocol * 64 + kc + 8 * kh);
      acc = wmma_f16(a, bm, acc);
    }
    float bias = b3[ocol];
#pragma unroll
    for (int v = 0; v < 8; ++v)
      h3[(mt * 16 + v + 8 * kh) * 64 + ocol] = fmaxf(acc[v] + bias, 0.f);
  }
  __syncthreads();

  // max over 10 neighbors (all in-block): 8 points x 64 ch
  for (int e = t; e < 8 * 64; e += 256) {
    int p = e >> 6, c = e & 63;
    float m = 0.f;   // relu outputs >= 0
#pragma unroll
    for (int k = 0; k < KNB; ++k) m = fmaxf(m, h3[(p * KNB + k) * 64 + c]);
    vert[(size_t)(pbase + p) * DD + c] = m;
  }
}

// ---------------- 8) global max over N (uint atomicMax, relu => >=0) ----------------
__global__ void k_gmax(const float* __restrict__ vert, unsigned int* __restrict__ g) {
  __shared__ float red[4][64];
  int t = threadIdx.x;
  int c = t & 63, rg = t >> 6;
  int rowbase = blockIdx.x * 64;
  float m = 0.f;
  for (int rr = rg; rr < 64; rr += 4) m = fmaxf(m, vert[(size_t)(rowbase + rr) * DD + c]);
  red[rg][c] = m;
  __syncthreads();
  if (t < 64) {
    float mm = fmaxf(fmaxf(red[0][t], red[1][t]), fmaxf(red[2][t], red[3][t]));
    atomicMax(&g[(rowbase >> 12) * 64 + t], __float_as_uint(mm));
  }
}

// ---------------- 9) final: [64x128]@[128x64] WMMA + relu ----------------
#define FLD 136
__global__ void __launch_bounds__(256) k_final(const float* __restrict__ vert,
                       const unsigned int* __restrict__ g,
                       const _Float16* __restrict__ wtg, const float* __restrict__ bg,
                       float* __restrict__ out) {
  extern __shared__ char smem[];
  _Float16* A = (_Float16*)smem;   // 64 x FLD
  int t = threadIdx.x;
  int lane = t & 31, wave = t >> 5;
  int kh = lane >> 4, l15 = lane & 15;
  int rowbase = blockIdx.x * 64;
  int b = rowbase >> 12;
  for (int e = t; e < 64 * 128; e += 256) {
    int row = e >> 7, col = e & 127;
    float v = (col < 64) ? vert[(size_t)(rowbase + row) * DD + col]
                         : __uint_as_float(g[b * 64 + (col - 64)]);
    A[row * FLD + col] = (_Float16)v;
  }
  __syncthreads();
  for (int tile = wave; tile < 16; tile += 8) {   // 4 mtiles x 4 ntiles
    int mt = tile >> 2, nt = tile & 3;
    int ocol = nt * 16 + l15;
    v8f acc = {};
#pragma unroll
    for (int kc = 0; kc < 128; kc += 32) {
      v16h a  = loadA16(A + (mt * 16 + l15) * FLD + kc + 8 * kh);
      v16h bm = loadB16(wtg + ocol * 128 + kc + 8 * kh);
      acc = wmma_f16(a, bm, acc);
    }
    float bias = bg[ocol];
#pragma unroll
    for (int v = 0; v < 8; ++v)
      out[(size_t)(rowbase + mt * 16 + v + 8 * kh) * DD + ocol] = fmaxf(acc[v] + bias, 0.f);
  }
}

// ---------------- launch ----------------
extern "C" void kernel_launch(void* const* d_in, const int* in_sizes, int n_in,
                              void* d_out, int out_size, void* d_ws, size_t ws_size,
                              hipStream_t stream) {
  (void)in_sizes; (void)n_in; (void)out_size; (void)ws_size;
  const float* x     = (const float*)d_in[0];
  const float* gamma = (const float*)d_in[1];
  const float* beta  = (const float*)d_in[2];
  const float* W1    = (const float*)d_in[3];
  const float* b1    = (const float*)d_in[4];
  const float* W2    = (const float*)d_in[5];
  const float* b2    = (const float*)d_in[6];
  const float* W3    = (const float*)d_in[7];
  const float* b3    = (const float*)d_in[8];
  const float* Wg    = (const float*)d_in[9];
  const float* bg    = (const float*)d_in[10];
  float* outp = (float*)d_out;

  char* ws = (char*)d_ws;
  float*        xn      = (float*)(ws + 0);               // 8,388,608 B
  float*        sqv     = (float*)(ws + 8388608);         //   131,072 B
  float*        stats   = (float*)(ws + 8519680);         //       512 B
  float*        partial = (float*)(ws + 8520192);         //   131,072 B
  int*          idx     = (int*)  (ws + 8651264);         // 1,310,720 B
  float*        vert    = (float*)(ws + 9961984);         // 8,388,608 B
  unsigned int* gbuf    = (unsigned int*)(ws + 18350592); //     2,048 B
  _Float16*     wt1     = (_Float16*)(ws + 18352640);     //    32,768 B
  _Float16*     wt2     = (_Float16*)(ws + 18385408);     //    16,384 B
  _Float16*     wt3     = (_Float16*)(ws + 18401792);     //     8,192 B
  _Float16*     wtg     = (_Float16*)(ws + 18409984);     //    16,384 B

  k_prep<<<144, 256, 0, stream>>>(W1, W2, W3, Wg, wt1, wt2, wt3, wtg);
  k_zero<<<2, 256, 0, stream>>>(gbuf);
  k_bnstats<<<256, 256, 0, stream>>>(x, partial);
  k_bnreduce<<<1, 128, 0, stream>>>(partial, stats);
  k_normalize<<<ROWS / 256, 256, 0, stream>>>(x, gamma, beta, stats, xn, sqv);

  size_t knn_smem = (size_t)(16 * AQLD + 16 + 16 * DLD + 16 * SEG * KNB * 2
                             + 16 * KNB * 2 + KCH * KLD) * 4;   // 59,200 B
  k_knn<<<BB * (NN / 16), 256, knn_smem, stream>>>(xn, sqv, idx);

  size_t mlp_smem = (size_t)80 * ELD * 2 + (size_t)80 * ELD * 2
                  + (size_t)80 * HLD * 2 + 80 * 4;              // 55,360 B
  k_mlp<<<ROWS / 8, 256, mlp_smem, stream>>>(xn, idx, wt1, wt2, wt3, b1, b2, b3, vert);

  k_gmax<<<ROWS / 64, 256, 0, stream>>>(vert, gbuf);
  k_final<<<ROWS / 64, 256, (size_t)64 * FLD * 2, stream>>>(vert, gbuf, wtg, bg, outp);
}
